// RoPEMultiHeadCrossAttention3D_8735963480745
// MI455X (gfx1250) — matrix operand order repaired and checked by reading, not compiled, not run on hardware
//
#include <hip/hip_runtime.h>
#include <hip/hip_bf16.h>

// ---------------------------------------------------------------------------
// RoPE multi-head cross-attention, MI455X (gfx1250, wave32, WMMA).
//
// fp32 inputs -> f16 Q/K/V with fp32 WMMA accumulation.
// head_dim = 32 == K of v_wmma_f32_16x16x32_f16 -> one WMMA per 16q x 16k
// logit tile. Flash-style streaming softmax: the 8*4096^2 attention matrix
// never touches HBM (512MB avoided; total traffic ~25MB @ 23.3TB/s ~ 1us).
// Weights are pre-converted to f16 once (0.64MB) so GEMM inner loops are
// pure load+WMMA with no per-iteration cvt chains. K/V tiles are staged
// into LDS with GLOBAL_LOAD_ASYNC_TO_LDS_B128 (ASYNCcnt path).
// ---------------------------------------------------------------------------

typedef _Float16 v16h __attribute__((ext_vector_type(16)));
typedef _Float16 h8   __attribute__((ext_vector_type(8)));
typedef float    v8f  __attribute__((ext_vector_type(8)));
typedef int      i4v  __attribute__((ext_vector_type(4)));   // async-LDS builtin arg type

#define NTOK   4096
#define CCH    256
#define NHEAD  8
#define HD     32

#if defined(__HIP_DEVICE_COMPILE__) && \
    __has_builtin(__builtin_amdgcn_global_load_async_to_lds_b128) && \
    __has_builtin(__builtin_amdgcn_s_wait_asynccnt)
#define USE_ASYNC_LDS 1
#endif

__device__ __forceinline__ v8f wmma16(v16h a, v16h b, v8f c) {
  // D = A(16x32 f16) * B(32x16 f16) + C(16x16 f32)
  return __builtin_amdgcn_wmma_f32_16x16x32_f16(false, a, false, b,
                                                (short)0, c, false, false);
}

// A-matrix fragment (16x32 f16), ISA 7.12.2:
//  lanes 0-15 : row M=lane,    halves = K[0..7]  then K[16..23]
//  lanes16-31 : row M=lane-16, halves = K[8..15] then K[24..31]
__device__ __forceinline__ v16h frag_a_h(const _Float16* base, int rs, int kbase) {
  int lane = threadIdx.x & 31;
  int r = lane & 15, hi = lane >> 4;
  const _Float16* p = base + r * rs + kbase + hi * 8;
  h8 lo = *(const h8*)(p);
  h8 hh = *(const h8*)(p + 16);
  v16h o;
#pragma unroll
  for (int i = 0; i < 8; ++i) { o[i] = lo[i]; o[i + 8] = hh[i]; }
  return o;
}

// B-matrix fragment (32x16 f16), ISA 7.12.2:
//  lanes 0-15 : col N=lane,    halves = K[0..7]  then K[16..23]
//  lanes16-31 : col N=lane-16, halves = K[8..15] then K[24..31]
// Source stored [col][K] with row stride `rs` halfs.
__device__ __forceinline__ v16h frag_b_h(const _Float16* base, int rs, int kbase) {
  int lane = threadIdx.x & 31;
  int c = lane & 15, hi = lane >> 4;
  const _Float16* p = base + c * rs + kbase + hi * 8;
  h8 lo = *(const h8*)(p);
  h8 hh = *(const h8*)(p + 16);
  v16h o;
#pragma unroll
  for (int i = 0; i < 8; ++i) { o[i] = lo[i]; o[i + 8] = hh[i]; }
  return o;
}

// ---------------------------------------------------------------------------
// Kernel 0: one-shot weight conversion f32 -> f16 (wq|wk|wv|wo).
// ---------------------------------------------------------------------------
__global__ __launch_bounds__(256)
void wcvt_kernel(const float* __restrict__ wq, const float* __restrict__ wk,
                 const float* __restrict__ wv, const float* __restrict__ wo,
                 _Float16* __restrict__ Wq, _Float16* __restrict__ Wk,
                 _Float16* __restrict__ Wv, _Float16* __restrict__ Wo) {
  const int which = blockIdx.y;
  const float* src = (which == 0) ? wq : (which == 1) ? wk : (which == 2) ? wv : wo;
  _Float16* dst = (which == 0) ? Wq : (which == 1) ? Wk : (which == 2) ? Wv : Wo;
  const int count = (which == 3) ? 2 * CCH * CCH : CCH * CCH;
  const int i = blockIdx.x * 256 + threadIdx.x;
  if (i < count) dst[i] = (_Float16)src[i];
}

// ---------------------------------------------------------------------------
// Kernel 1: QKV projection + RoPE.  grid = (NTOK/16, 3 projections)
//   out[co,n] = sum_ci W[co,ci] * x[ci,n] + b[co], RoPE on q/k.
//   Q,K stored token-major f16 [h][tok][32]; V channel-major [h][32][tok].
// ---------------------------------------------------------------------------
__global__ __launch_bounds__(256)
void qkv_rope_kernel(const float* __restrict__ x,
                     const _Float16* __restrict__ Wq, const float* __restrict__ bq,
                     const _Float16* __restrict__ Wk, const float* __restrict__ bk,
                     const _Float16* __restrict__ Wv, const float* __restrict__ bv,
                     _Float16* __restrict__ Qb, _Float16* __restrict__ Kb,
                     _Float16* __restrict__ Vb) {
  __shared__ __align__(16) _Float16 Xs[16 * CCH];   // 16 tokens x 256 ci
  const int tokbase = blockIdx.x * 16;
  const int proj = blockIdx.y;                       // 0=q, 1=k, 2=v

  const _Float16* w = (proj == 0) ? Wq : (proj == 1) ? Wk : Wv;
  const float* bias = (proj == 0) ? bq : (proj == 1) ? bk : bv;
  const int xoff = (proj == 0) ? 0 : CCH * NTOK;     // q reads first half of x

  for (int i = threadIdx.x; i < 16 * CCH; i += 256) {
    int t = i >> 8, c = i & 255;
    Xs[t * CCH + c] = (_Float16)x[xoff + c * NTOK + tokbase + t];
  }
  __syncthreads();

  const int wave = threadIdx.x >> 5;
  const int lane = threadIdx.x & 31;
  const int col = lane & 15, hi = lane >> 4;
  const int token = tokbase + col;
  const float pos = -1.f + 2.f * (float)token / (float)(NTOK - 1);

  for (int ct = wave; ct < CCH / 16; ct += 8) {      // 16 co-tiles over 8 waves
    const int co0 = ct * 16;
    v8f acc = {};
#pragma unroll
    for (int kk = 0; kk < CCH; kk += 32) {
      v16h a = frag_a_h(w + co0 * CCH, CCH, kk);     // f16 weights, pure load
      v16h b = frag_b_h(Xs, CCH, kk);
      acc = wmma16(a, b, acc);
    }
    // C layout: lane holds col N=token(L&15), rows M = r + 8*hi.
#pragma unroll
    for (int r = 0; r < 8; ++r) {
      const int gc = co0 + r + hi * 8;
      float val = acc[r] + bias[gc];
      const int ch = gc & (HD - 1);
      const int h  = gc >> 5;
      if (proj < 2) {                                // RoPE on q and k
        const int c2 = ch & 15;
        const float invf = __powf(10000.f, -(float)c2 / 16.f);
        const float ang = pos * invf;
        val *= (ch < 16) ? __sinf(ang) : __cosf(ang);
      }
      const _Float16 hv = (_Float16)val;
      if (proj == 0)      Qb[((size_t)h * NTOK + token) * HD + ch] = hv;
      else if (proj == 1) Kb[((size_t)h * NTOK + token) * HD + ch] = hv;
      else                Vb[((size_t)h * HD + ch) * NTOK + token] = hv;
    }
  }
}

// ---------------------------------------------------------------------------
// Kernel 2: flash attention. grid = (NTOK/128, NHEAD), 256 thr = 8 waves.
// Per 32-key chunk: async-copy K/V tiles to LDS (GLOBAL_LOAD_ASYNC_TO_LDS,
// ASYNCcnt), 2 WMMA S=Q^T K, online softmax via half-wave shuffles,
// P transposed through LDS, 2 WMMA O += P V^T. Epilogue fuses +qf residual.
// ---------------------------------------------------------------------------
__global__ __launch_bounds__(256)
void flash_attn_kernel(const _Float16* __restrict__ Qb,
                       const _Float16* __restrict__ Kb,
                       const _Float16* __restrict__ Vb,
                       const float* __restrict__ x,
                       _Float16* __restrict__ U) {
  __shared__ __align__(16) _Float16 Ks[32 * 32];     // [key][ch]
  __shared__ __align__(16) _Float16 Vs[32 * 32];     // [ch][key]
  __shared__ __align__(16) _Float16 Ps[8][16 * 32];  // per-wave P tile [q][key]

  const int h = blockIdx.y;
  const int wave = threadIdx.x >> 5, lane = threadIdx.x & 31;
  const int col = lane & 15, hi = lane >> 4;
  const int qbase = blockIdx.x * 128 + wave * 16;

  const _Float16* Qh = Qb + (size_t)h * NTOK * HD;
  const _Float16* Kh = Kb + (size_t)h * NTOK * HD;
  const _Float16* Vh = Vb + (size_t)h * HD * NTOK;

  const v16h aq = frag_a_h(Qh + (size_t)qbase * HD, HD, 0);

  v8f o0 = {}, o1 = {};
  float m[8], l[8];
#pragma unroll
  for (int r = 0; r < 8; ++r) { m[r] = -1e30f; l[r] = 0.f; }

  // Per-thread 16B chunk assignment for the 2KB K and V tiles (128+128 chunks).
  const int tid = threadIdx.x;
  const int crow = (tid & 127) >> 2;                 // row within tile (0..31)
  const int coff = (tid & 3) * 8;                    // half-offset within row

  for (int kb = 0; kb < NTOK; kb += 32) {
    __syncthreads();                                 // tile consumers done
#ifdef USE_ASYNC_LDS
    if (tid < 128) {                                 // K tile: [key][ch]
      __builtin_amdgcn_global_load_async_to_lds_b128(
          (i4v*)(Kh + (size_t)(kb + crow) * HD + coff),
          (i4v*)(Ks + crow * 32 + coff), 0, 0);
    } else {                                         // V tile: [ch][key]
      __builtin_amdgcn_global_load_async_to_lds_b128(
          (i4v*)(Vh + (size_t)crow * NTOK + kb + coff),
          (i4v*)(Vs + crow * 32 + coff), 0, 0);
    }
    __builtin_amdgcn_s_wait_asynccnt(0);
#else
    if (tid < 128) {
      *(h8*)(Ks + crow * 32 + coff) = *(const h8*)(Kh + (size_t)(kb + crow) * HD + coff);
    } else {
      *(h8*)(Vs + crow * 32 + coff) = *(const h8*)(Vh + (size_t)crow * NTOK + kb + coff);
    }
#endif
    if (kb + 32 < NTOK)                              // CDNA5 global_prefetch
      __builtin_prefetch(Kh + (size_t)(kb + 32) * HD + (threadIdx.x << 2), 0, 1);
    __syncthreads();

    const v16h bk0 = frag_b_h(Ks, 32, 0);            // keys kb+0..15
    const v16h bk1 = frag_b_h(Ks + 16 * 32, 32, 0);  // keys kb+16..31
    v8f z = {};
    v8f s0 = wmma16(aq, bk0, z);
    v8f s1 = wmma16(aq, bk1, z);

    // Online softmax; row stats per half-wave: row = r + 8*hi.
#pragma unroll
    for (int r = 0; r < 8; ++r) {
      float t = fmaxf(s0[r], s1[r]);
      t = fmaxf(t, __shfl_xor(t, 1));
      t = fmaxf(t, __shfl_xor(t, 2));
      t = fmaxf(t, __shfl_xor(t, 4));
      t = fmaxf(t, __shfl_xor(t, 8));
      const float mn = fmaxf(m[r], t);
      const float sc = __expf(m[r] - mn);
      const float p0 = __expf(s0[r] - mn);
      const float p1 = __expf(s1[r] - mn);
      float u = p0 + p1;
      u += __shfl_xor(u, 1); u += __shfl_xor(u, 2);
      u += __shfl_xor(u, 4); u += __shfl_xor(u, 8);
      l[r] = l[r] * sc + u;
      m[r] = mn;
      o0[r] *= sc; o1[r] *= sc;
      const int row = r + hi * 8;                    // C-layout -> [q][key]
      Ps[wave][row * 32 + col]      = (_Float16)p0;
      Ps[wave][row * 32 + 16 + col] = (_Float16)p1;
    }
    __syncthreads();

    const v16h ap  = frag_a_h(Ps[wave], 32, 0);      // P: 16 q x 32 keys
    const v16h bv0 = frag_b_h(Vs, 32, 0);            // channels 0..15
    const v16h bv1 = frag_b_h(Vs + 16 * 32, 32, 0);  // channels 16..31
    o0 = wmma16(ap, bv0, o0);
    o1 = wmma16(ap, bv1, o1);
  }

  // Normalize, add qf residual, store U token-major f16 [tok][256].
#pragma unroll
  for (int r = 0; r < 8; ++r) {
    const int token = qbase + r + hi * 8;
    const float inv = 1.f / l[r];
    const int c0 = h * HD + col;
    const int c1 = h * HD + 16 + col;
    U[(size_t)token * CCH + c0] = (_Float16)(o0[r] * inv + x[(size_t)c0 * NTOK + token]);
    U[(size_t)token * CCH + c1] = (_Float16)(o1[r] * inv + x[(size_t)c1 * NTOK + token]);
  }
}

// ---------------------------------------------------------------------------
// Kernel 3: output projection. y[o,n] = Wo[o,:]·U[n,:] + bo[o], o < 512.
// ---------------------------------------------------------------------------
__global__ __launch_bounds__(256)
void oproj_kernel(const _Float16* __restrict__ U,
                  const _Float16* __restrict__ Wo, const float* __restrict__ bo,
                  float* __restrict__ out) {
  __shared__ __align__(16) _Float16 Us[16 * CCH];
  const int tokbase = blockIdx.x * 16;
  for (int i = threadIdx.x; i < 16 * CCH; i += 256)
    Us[i] = U[(size_t)(tokbase + (i >> 8)) * CCH + (i & 255)];
  __syncthreads();

  const int wave = threadIdx.x >> 5, lane = threadIdx.x & 31;
  const int col = lane & 15, hi = lane >> 4;
  const int token = tokbase + col;

  for (int ot = wave; ot < (2 * CCH) / 16; ot += 8) {
    const int oc0 = ot * 16;
    v8f acc = {};
#pragma unroll
    for (int kk = 0; kk < CCH; kk += 32) {
      v16h a = frag_a_h(Wo + (size_t)oc0 * CCH, CCH, kk);
      v16h b = frag_b_h(Us, CCH, kk);
      acc = wmma16(a, b, acc);
    }
#pragma unroll
    for (int r = 0; r < 8; ++r) {
      const int oc = oc0 + r + hi * 8;
      out[(size_t)oc * NTOK + token] = acc[r] + bo[oc];
    }
  }
}

// ---------------------------------------------------------------------------
extern "C" void kernel_launch(void* const* d_in, const int* in_sizes, int n_in,
                              void* d_out, int out_size, void* d_ws, size_t ws_size,
                              hipStream_t stream) {
  const float* x  = (const float*)d_in[0];
  const float* wq = (const float*)d_in[1];
  const float* bq = (const float*)d_in[2];
  const float* wk = (const float*)d_in[3];
  const float* bk = (const float*)d_in[4];
  const float* wv = (const float*)d_in[5];
  const float* bv = (const float*)d_in[6];
  const float* wo = (const float*)d_in[7];
  const float* bo = (const float*)d_in[8];
  float* out = (float*)d_out;

  // Workspace (f16): Q,K,V = 8*4096*32 each, U = 4096*256, then f16 weights.
  _Float16* Qb = (_Float16*)d_ws;
  _Float16* Kb = Qb + (size_t)NHEAD * NTOK * HD;
  _Float16* Vb = Kb + (size_t)NHEAD * NTOK * HD;
  _Float16* U  = Vb + (size_t)NHEAD * NTOK * HD;
  _Float16* Wq = U  + (size_t)NTOK * CCH;
  _Float16* Wk = Wq + (size_t)CCH * CCH;
  _Float16* Wv = Wk + (size_t)CCH * CCH;
  _Float16* Wo = Wv + (size_t)CCH * CCH;

  wcvt_kernel<<<dim3(512, 4), 256, 0, stream>>>(wq, wk, wv, wo, Wq, Wk, Wv, Wo);
  qkv_rope_kernel<<<dim3(NTOK / 16, 3), 256, 0, stream>>>(
      x, Wq, bq, Wk, bk, Wv, bv, Qb, Kb, Vb);
  flash_attn_kernel<<<dim3(NTOK / 128, NHEAD), 256, 0, stream>>>(
      Qb, Kb, Vb, x, U);
  oproj_kernel<<<NTOK / 16, 256, 0, stream>>>(U, Wo, bo, out);
}
